// LTX2Attention_65859028517578
// MI455X (gfx1250) — compile-verified, matrix operand order checked by reading, not run on hardware
//
#include <hip/hip_runtime.h>
#include <hip/hip_bf16.h>
#include <math.h>

// ---------------------------------------------------------------------------
// Problem constants (match reference)
// ---------------------------------------------------------------------------
#define BQ   2
#define SQ   2048
#define QD   4096          // Q_DIM
#define NH   32
#define HD   128
#define INNER 4096         // NH*HD
#define MROWS 4096         // B*S
#define EPSV 1e-6f

typedef __bf16 bf16;
typedef bf16  v16bf __attribute__((ext_vector_type(16)));
typedef bf16  v8bf  __attribute__((ext_vector_type(8)));
typedef float v8f   __attribute__((ext_vector_type(8)));
typedef unsigned int uint32x4 __attribute__((ext_vector_type(4)));
typedef int int32x4 __attribute__((ext_vector_type(4)));
typedef int int32x8 __attribute__((ext_vector_type(8)));
typedef int v4i __attribute__((ext_vector_type(4)));
typedef __attribute__((address_space(1))) v4i* global_v4i_ptr;
typedef __attribute__((address_space(3))) v4i* lds_v4i_ptr;

// ---------------------------------------------------------------------------
// CDNA5 async / tensor data movement helpers (guarded; asm fallback)
// ---------------------------------------------------------------------------
#if defined(__has_builtin)
#if __has_builtin(__builtin_amdgcn_global_load_async_to_lds_b128)
#define HAVE_ASYNC_BUILTIN 1
#endif
#if __has_builtin(__builtin_amdgcn_tensor_load_to_lds)
#define HAVE_TDM_BUILTIN 1
#endif
#if __has_builtin(__builtin_amdgcn_s_wait_asynccnt)
#define HAVE_WAIT_ASYNC_BUILTIN 1
#endif
#if __has_builtin(__builtin_amdgcn_s_wait_tensorcnt)
#define HAVE_WAIT_TENSOR_BUILTIN 1
#endif
#endif

// 16-byte async copy: global -> LDS, tracked by ASYNCcnt.
__device__ __forceinline__ void async_copy16(void* lds_dst, const void* gsrc) {
#if defined(HAVE_ASYNC_BUILTIN)
  __builtin_amdgcn_global_load_async_to_lds_b128(
      (global_v4i_ptr)(uintptr_t)gsrc,
      (lds_v4i_ptr)(unsigned)(uintptr_t)lds_dst, 0, 0);
#else
  const unsigned ldsa = (unsigned)(uintptr_t)lds_dst;
  asm volatile("global_load_async_to_lds_b128 %0, %1, off" ::"v"(ldsa),
               "v"(gsrc)
               : "memory");
#endif
}

__device__ __forceinline__ void wait_async0() {
#if defined(HAVE_WAIT_ASYNC_BUILTIN)
  __builtin_amdgcn_s_wait_asynccnt(0);
#else
  asm volatile("s_wait_asynccnt 0x0" ::: "memory");
#endif
}

__device__ __forceinline__ void wait_tensor0() {
#if defined(HAVE_WAIT_TENSOR_BUILTIN)
  __builtin_amdgcn_s_wait_tensorcnt(0);
#else
  asm volatile("s_wait_tensorcnt 0x0" ::: "memory");
#endif
}

#if defined(HAVE_TDM_BUILTIN)
// TDM load of a 2-D bf16 tile [tile_d1 rows x tile_d0 elems], row stride
// stride0 elems, from global to LDS.  Builds D# groups 0/1 per ISA 8.3/8.4
// (count=1, type=2 "image", data_size code 1 = 2 bytes, tensor dims == tile).
__device__ __forceinline__ void tdm_load_tile_bf16(unsigned lds_off,
                                                   const void* gaddr,
                                                   unsigned tile_d0,
                                                   unsigned tile_d1,
                                                   unsigned stride0) {
  const unsigned long long ga = (unsigned long long)(uintptr_t)gaddr;
  uint32x4 g0;
  g0[0] = 1u;                                   // count=1, user descriptor
  g0[1] = lds_off;                              // lds_addr
  g0[2] = (unsigned)(ga & 0xFFFFFFFFu);         // global_addr[31:0]
  g0[3] = (unsigned)((ga >> 32) & 0x01FFFFFFu) | (2u << 30);  // [56:32]|type=2
  int32x8 g1;
  g1[0] = (int)(1u << 16);                      // data_size=1 (2B), mask=0
  g1[1] = (int)((tile_d0 & 0xFFFFu) << 16);     // tensor_dim0[15:0]
  g1[2] = (int)(((tile_d0 >> 16) & 0xFFFFu) | ((tile_d1 & 0xFFFFu) << 16));
  g1[3] = (int)(((tile_d1 >> 16) & 0xFFFFu) | ((tile_d0 & 0xFFFFu) << 16));
  g1[4] = (int)(tile_d1 & 0xFFFFu);             // tile_dim1 (tile_dim2=0)
  g1[5] = (int)stride0;                         // tensor_dim0_stride[31:0]
  g1[6] = 0;
  g1[7] = 0;
  const int32x4 z4 = {0, 0, 0, 0};
#if __clang_major__ >= 23
  const int32x8 z8 = {0, 0, 0, 0, 0, 0, 0, 0};
  __builtin_amdgcn_tensor_load_to_lds(g0, g1, z4, z4, z8, 0);
#else
  __builtin_amdgcn_tensor_load_to_lds(g0, g1, z4, z4, 0);
#endif
}
#endif

// ---------------------------------------------------------------------------
// WMMA helper: D = A(16x32 bf16) * B(32x16 bf16) + C(16x16 f32)
// ---------------------------------------------------------------------------
__device__ __forceinline__ v8f wmma_bf16(v16bf a, v16bf b, v8f c) {
  return __builtin_amdgcn_wmma_f32_16x16x32_bf16(
      /*neg_a=*/false, a, /*neg_b=*/false, b,
      /*c_mod=*/(short)0, c, /*reuse_a=*/false, /*reuse_b=*/false);
}

// Load a 16x32 bf16 fragment (A layout; B uses same per-lane mapping when the
// source is stored transposed, i.e. [n][k]).  Per ISA 7.12.2:
//   lanes 0-15 : VGPR0-3 -> K 0..7,  VGPR4-7 -> K 16..23
//   lanes 16-31: VGPR0-3 -> K 8..15, VGPR4-7 -> K 24..31
__device__ __forceinline__ v16bf load_frag(const bf16* base, int stride) {
  const int lane = threadIdx.x & 31;
  const bf16* p = base + (size_t)(lane & 15) * stride + ((lane >> 4) << 3);
  v8bf lo = *reinterpret_cast<const v8bf*>(p);
  v8bf hi = *reinterpret_cast<const v8bf*>(p + 16);
  return __builtin_shufflevector(lo, hi, 0, 1, 2, 3, 4, 5, 6, 7,
                                          8, 9, 10, 11, 12, 13, 14, 15);
}

// ---------------------------------------------------------------------------
// GEMM: C[4096,4096] = A @ W + bias (A fp32 or bf16; W/bias fp32 -> bf16 in
// the loader).  mode 0: fp32 row-major out; mode 1: bf16 [B,H,S,D] out.
// ---------------------------------------------------------------------------
template <bool A_IS_BF16>
__global__ __launch_bounds__(256) void gemm_wmma_kernel(
    const void* __restrict__ Aptr, const float* __restrict__ W,
    const float* __restrict__ bias, void* __restrict__ Cout, int mode) {
  __shared__ __align__(16) bf16 ldsA[128 * 32];   // [m][k]
  __shared__ __align__(16) bf16 ldsB[128 * 32];   // [n][k]  (transposed)

  const int tid  = threadIdx.x;
  const int lane = tid & 31;
  const int wave = tid >> 5;
  const int m0 = blockIdx.y * 128;
  const int n0 = blockIdx.x * 128;
  const int wm = (wave & 3) * 32;
  const int wn = (wave >> 2) * 64;

  v8f acc[2][4];
#pragma unroll
  for (int i = 0; i < 2; ++i)
#pragma unroll
    for (int j = 0; j < 4; ++j) acc[i][j] = (v8f)(0.0f);

  const int arow = tid >> 1;          // 0..127
  const int acol = (tid & 1) * 16;    // 0 or 16
  const int brow = tid >> 3;          // 0..31 (k)
  const int bcol = (tid & 7) * 16;    // 0..112 step 16 (n)

  for (int k0 = 0; k0 < QD; k0 += 32) {
    __syncthreads();  // previous iteration's LDS reads complete

    // ---- stage A tile ----
    if constexpr (A_IS_BF16) {
      const bf16* A = (const bf16*)Aptr;
      const bf16* src = A + (size_t)(m0 + arow) * QD + k0 + acol;
      async_copy16(&ldsA[arow * 32 + acol], src);       // ASYNCcnt-tracked
      async_copy16(&ldsA[arow * 32 + acol + 8], src + 8);
    } else {
      const float* A = (const float*)Aptr;
      const float* src = A + (size_t)(m0 + arow) * QD + k0 + acol;
      bf16* dst = &ldsA[arow * 32 + acol];
#pragma unroll
      for (int i = 0; i < 16; ++i) dst[i] = (bf16)src[i];
    }

    // ---- stage B tile transposed: ldsB[n][k] (fp32 -> bf16) ----
    const float* bsrc = W + (size_t)(k0 + brow) * INNER + n0 + bcol;
#pragma unroll
    for (int i = 0; i < 16; ++i) ldsB[(bcol + i) * 32 + brow] = (bf16)bsrc[i];

    if (k0 + 32 < QD) {
      __builtin_prefetch(bsrc + (size_t)32 * INNER, 0, 1);  // global_prefetch_b8
    }

    if constexpr (A_IS_BF16) wait_async0();
    __syncthreads();  // tiles visible

    // ---- compute: 2 (m) x 4 (n) WMMA tiles per wave ----
    v16bf afrag0 = load_frag(&ldsA[(wm + 0) * 32], 32);
    v16bf afrag1 = load_frag(&ldsA[(wm + 16) * 32], 32);
#pragma unroll
    for (int ni = 0; ni < 4; ++ni) {
      v16bf bfrag = load_frag(&ldsB[(wn + ni * 16) * 32], 32);
      acc[0][ni] = wmma_bf16(afrag0, bfrag, acc[0][ni]);
      acc[1][ni] = wmma_bf16(afrag1, bfrag, acc[1][ni]);
    }
  }

  // ---- epilogue ----
  const int half = lane >> 4;
  const int nlane = lane & 15;
#pragma unroll
  for (int mi = 0; mi < 2; ++mi)
#pragma unroll
    for (int ni = 0; ni < 4; ++ni)
#pragma unroll
      for (int r = 0; r < 8; ++r) {
        const int row = m0 + wm + mi * 16 + r + half * 8;
        const int col = n0 + wn + ni * 16 + nlane;
        const float v = acc[mi][ni][r] + bias[col];
        if (mode == 0) {
          ((float*)Cout)[(size_t)row * INNER + col] = v;
        } else {
          const int b = row >> 11, s = row & (SQ - 1);
          const int h = col >> 7, d = col & (HD - 1);
          ((bf16*)Cout)[(((size_t)b * NH + h) * SQ + s) * HD + d] = (bf16)v;
        }
      }
}

// ---------------------------------------------------------------------------
// In-place RMSNorm over INNER of a [B,H,S,D] bf16 tensor.
// ---------------------------------------------------------------------------
__global__ __launch_bounds__(256) void rmsnorm_bhsd_kernel(
    bf16* __restrict__ x, const float* __restrict__ g) {
  const int row = blockIdx.x;            // b*SQ + s
  const int b = row >> 11, s = row & (SQ - 1);
  const int tid = threadIdx.x;
  __shared__ float red[256];

  float vals[16];
  float acc = 0.0f;
#pragma unroll
  for (int i = 0; i < 16; ++i) {
    const int c = tid + i * 256;
    const int h = c >> 7, d = c & (HD - 1);
    const size_t idx = (((size_t)b * NH + h) * SQ + s) * HD + d;
    const float v = (float)x[idx];
    vals[i] = v;
    acc += v * v;
  }
  red[tid] = acc;
  __syncthreads();
  for (int off = 128; off > 0; off >>= 1) {
    if (tid < off) red[tid] += red[tid + off];
    __syncthreads();
  }
  const float scale = rsqrtf(red[0] * (1.0f / (float)INNER) + EPSV);
#pragma unroll
  for (int i = 0; i < 16; ++i) {
    const int c = tid + i * 256;
    const int h = c >> 7, d = c & (HD - 1);
    const size_t idx = (((size_t)b * NH + h) * SQ + s) * HD + d;
    x[idx] = (bf16)(vals[i] * scale * g[c]);
  }
}

// ---------------------------------------------------------------------------
// Flash attention: grid (B*H, S/128), block 256 (8 waves).  K tile staged via
// async-to-LDS b128; V tile staged row-major via TDM (when available) then
// transposed in LDS; online softmax in WMMA C layout.
// ---------------------------------------------------------------------------
__global__ __launch_bounds__(256) void attention_wmma_kernel(
    const bf16* __restrict__ Q, const bf16* __restrict__ K,
    const bf16* __restrict__ V, bf16* __restrict__ O) {
  __shared__ __align__(16) bf16 ldsK[32 * 128];    // [key][d]
  __shared__ __align__(16) bf16 ldsVr[32 * 128];   // [key][d] staging
  __shared__ __align__(16) bf16 ldsVt[128 * 32];   // [d][key]
  __shared__ __align__(16) bf16 ldsP[8 * 16 * 32]; // per-wave P tile

  const int bh = blockIdx.x;             // b*NH + h
  const int q0 = blockIdx.y * 128;
  const int tid = threadIdx.x;
  const int lane = tid & 31;
  const int wave = tid >> 5;
  const int half = lane >> 4;
  const int nlane = lane & 15;

  // resident Q fragments (16 rows x 128 d, as 4 K=32 A-fragments)
  const bf16* qbase = Q + ((size_t)bh * SQ + q0 + wave * 16) * HD;
  v16bf qfrag[4];
#pragma unroll
  for (int dc = 0; dc < 4; ++dc) qfrag[dc] = load_frag(qbase + dc * 32, HD);

  v8f o[8];
#pragma unroll
  for (int i = 0; i < 8; ++i) o[i] = (v8f)(0.0f);
  float mrow[8], lrow[8];
#pragma unroll
  for (int r = 0; r < 8; ++r) {
    mrow[r] = -__builtin_inff();
    lrow[r] = 0.0f;
  }

  const int krow = tid >> 3;          // 0..31
  const int kcol = (tid & 7) * 16;    // 0..112 step 16
  const float sscale = 0.088388347648318447f;  // 1/sqrt(128)

  for (int kt = 0; kt < SQ; kt += 32) {
    __syncthreads();  // previous iteration's LDS consumers are done

    // ---- stage K tile [key][d] via async global->LDS ----
    const bf16* ksrc = K + ((size_t)bh * SQ + kt + krow) * HD + kcol;
    async_copy16(&ldsK[krow * HD + kcol], ksrc);
    async_copy16(&ldsK[krow * HD + kcol + 8], ksrc + 8);

    // ---- stage V tile [key][d] via TDM (else async), then transpose ----
    const bf16* vtile = V + ((size_t)bh * SQ + kt) * HD;
#if defined(HAVE_TDM_BUILTIN)
    if (wave == 0) {
      tdm_load_tile_bf16((unsigned)(uintptr_t)&ldsVr[0], vtile, HD, 32, HD);
      wait_tensor0();
    }
#else
    {
      const bf16* vsrc = vtile + (size_t)krow * HD + kcol;
      async_copy16(&ldsVr[krow * HD + kcol], vsrc);
      async_copy16(&ldsVr[krow * HD + kcol + 8], vsrc + 8);
    }
#endif
    wait_async0();
    __syncthreads();  // ldsK + ldsVr ready

    // transpose V staging into [d][key]
#pragma unroll
    for (int i = 0; i < 16; ++i)
      ldsVt[(kcol + i) * 32 + krow] = ldsVr[krow * HD + kcol + i];
    __syncthreads();  // ldsVt ready

    // ---- scores: 16 q-rows x 32 keys (two 16x16 C tiles) ----
    v8f sc[2];
    sc[0] = (v8f)(0.0f);
    sc[1] = (v8f)(0.0f);
#pragma unroll
    for (int st = 0; st < 2; ++st)
#pragma unroll
      for (int dc = 0; dc < 4; ++dc) {
        v16bf kfrag = load_frag(&ldsK[(st * 16) * HD + dc * 32], HD);
        sc[st] = wmma_bf16(qfrag[dc], kfrag, sc[st]);
      }
#pragma unroll
    for (int st = 0; st < 2; ++st)
#pragma unroll
      for (int r = 0; r < 8; ++r) sc[st][r] *= sscale;

    // ---- online softmax stats (row = r + half*8) ----
    float mnew[8], alpha[8];
#pragma unroll
    for (int r = 0; r < 8; ++r) {
      float v = fmaxf(sc[0][r], sc[1][r]);
#pragma unroll
      for (int m = 1; m < 16; m <<= 1) v = fmaxf(v, __shfl_xor(v, m, 32));
      mnew[r] = fmaxf(mrow[r], v);
      alpha[r] = __expf(mrow[r] - mnew[r]);
      mrow[r] = mnew[r];
    }

#pragma unroll
    for (int st = 0; st < 2; ++st)
#pragma unroll
      for (int r = 0; r < 8; ++r) {
        const float p = __expf(sc[st][r] - mnew[r]);
        sc[st][r] = p;
        ldsP[wave * 512 + (r + half * 8) * 32 + st * 16 + nlane] = (bf16)p;
      }
#pragma unroll
    for (int r = 0; r < 8; ++r) {
      float v = sc[0][r] + sc[1][r];
#pragma unroll
      for (int m = 1; m < 16; m <<= 1) v += __shfl_xor(v, m, 32);
      lrow[r] = lrow[r] * alpha[r] + v;
    }
#pragma unroll
    for (int dt = 0; dt < 8; ++dt)
#pragma unroll
      for (int r = 0; r < 8; ++r) o[dt][r] *= alpha[r];

    __syncthreads();  // ldsP visible

    // ---- O += P(16x32) @ V(32x128) ----
    v16bf pfrag = load_frag(&ldsP[wave * 512], 32);
#pragma unroll
    for (int dt = 0; dt < 8; ++dt) {
      v16bf vfrag = load_frag(&ldsVt[dt * 16 * 32], 32);
      o[dt] = wmma_bf16(pfrag, vfrag, o[dt]);
    }
  }

  // writeout: attn_out[b*SQ + s][h*HD + d] bf16
  const int b = bh >> 5, h = bh & (NH - 1);
#pragma unroll
  for (int dt = 0; dt < 8; ++dt)
#pragma unroll
    for (int r = 0; r < 8; ++r) {
      const int srow = q0 + wave * 16 + r + half * 8;
      const float val = o[dt][r] / lrow[r];
      O[((size_t)b * SQ + srow) * INNER + h * HD + dt * 16 + nlane] = (bf16)val;
    }
}

// ---------------------------------------------------------------------------
// Launch
// ---------------------------------------------------------------------------
extern "C" void kernel_launch(void* const* d_in, const int* in_sizes, int n_in,
                              void* d_out, int out_size, void* d_ws,
                              size_t ws_size, hipStream_t stream) {
  (void)in_sizes; (void)n_in; (void)out_size; (void)ws_size;
  const float* hs = (const float*)d_in[0];
  const float* Wq = (const float*)d_in[1];
  const float* bq = (const float*)d_in[2];
  const float* Wk = (const float*)d_in[3];
  const float* bk = (const float*)d_in[4];
  const float* Wv = (const float*)d_in[5];
  const float* bv = (const float*)d_in[6];
  const float* Wo = (const float*)d_in[7];
  const float* bo = (const float*)d_in[8];
  const float* gq = (const float*)d_in[9];
  const float* gk = (const float*)d_in[10];

  const size_t SZ = (size_t)MROWS * INNER * sizeof(bf16);  // 32 MiB
  char* ws = (char*)d_ws;
  bf16* Qb = (bf16*)(ws + 0 * SZ);   // [B,H,S,D]
  bf16* Kb = (bf16*)(ws + 1 * SZ);   // [B,H,S,D]
  bf16* Vb = (bf16*)(ws + 2 * SZ);   // [B,H,S,D]
  bf16* Ab = (bf16*)(ws + 3 * SZ);   // [B*S, INNER]

  dim3 gGemm(INNER / 128, MROWS / 128);
  gemm_wmma_kernel<false><<<gGemm, 256, 0, stream>>>(hs, Wq, bq, Qb, 1);
  gemm_wmma_kernel<false><<<gGemm, 256, 0, stream>>>(hs, Wk, bk, Kb, 1);
  gemm_wmma_kernel<false><<<gGemm, 256, 0, stream>>>(hs, Wv, bv, Vb, 1);

  rmsnorm_bhsd_kernel<<<MROWS, 256, 0, stream>>>(Qb, gq);
  rmsnorm_bhsd_kernel<<<MROWS, 256, 0, stream>>>(Kb, gk);

  attention_wmma_kernel<<<dim3(BQ * NH, SQ / 128), 256, 0, stream>>>(Qb, Kb, Vb,
                                                                     Ab);

  gemm_wmma_kernel<true><<<gGemm, 256, 0, stream>>>(Ab, Wo, bo, d_out, 0);
}
